// DIRNetDeform_86423331930423
// MI455X (gfx1250) — compile-verified, matrix operand order checked by reading, not run on hardware
//
#include <hip/hip_runtime.h>
#include <hip/hip_bf16.h>

typedef __attribute__((ext_vector_type(16))) _Float16 v16h;
typedef __attribute__((ext_vector_type(8)))  _Float16 v8h;
typedef __attribute__((ext_vector_type(8)))  float    v8f;

#define CHUNK_M 16384
#define KPAD_MAX 1152

// ---------------------------------------------------------------------------
// Weight prep: w (OIHW, f32) -> Bt f16 [Npad][Kpad], Bt[n][c*KH*KW+ky*KW+kx].
// transposed=1: w is (I,O,kh,kw); use w[c][n][KH-1-ky][KW-1-kx] (convT).
// ---------------------------------------------------------------------------
__global__ void k_prep_wB(const float* __restrict__ w, _Float16* __restrict__ Bt,
                          int Cout, int Cin, int KH, int KW, int Kpad, int Npad,
                          int transposed)
{
    long long idx = (long long)blockIdx.x * blockDim.x + threadIdx.x;
    long long total = (long long)Npad * Kpad;
    if (idx >= total) return;
    int k = (int)(idx % Kpad);
    int n = (int)(idx / Kpad);
    _Float16 val = (_Float16)0.f;
    int K = Cin * KH * KW;
    if (n < Cout && k < K) {
        int kx = k % KW;
        int t  = k / KW;
        int ky = t % KH;
        int c  = t / KH;
        float wv;
        if (!transposed)
            wv = w[(((size_t)n * Cin + c) * KH + ky) * KW + kx];
        else
            wv = w[(((size_t)c * Cout + n) * KH + (KH - 1 - ky)) * KW + (KW - 1 - kx)];
        val = (_Float16)wv;
    }
    Bt[(size_t)n * Kpad + k] = val;
}

// ---------------------------------------------------------------------------
// Generic im2col into f16 [Mc][Kpad].  Virtual input = X dilated by `dil`
// (lhs dilation, for transposed conv) with border pad `pad`.  stride==1.
// Arbitrary element strides so it reads NCHW (x1) or padded NHWC.
// ---------------------------------------------------------------------------
__global__ void k_im2col(const float* __restrict__ X, _Float16* __restrict__ col,
                         int m0, int Mc, int H, int W, int Cin,
                         long long sN, long long sC, long long sH, long long sW,
                         int OH, int OW, int KH, int KW, int pad, int dil, int Kpad)
{
    long long idx = (long long)blockIdx.x * blockDim.x + threadIdx.x;
    long long total = (long long)Mc * Kpad;
    if (idx >= total) return;
    int k  = (int)(idx % Kpad);
    int mr = (int)(idx / Kpad);
    int m  = m0 + mr;
    int ox = m % OW;
    int t  = m / OW;
    int oy = t % OH;
    int b  = t / OH;
    _Float16 val = (_Float16)0.f;
    int K = Cin * KH * KW;
    if (k < K) {
        int kx = k % KW;
        int t2 = k / KW;
        int ky = t2 % KH;
        int c  = t2 / KH;
        int u = oy + ky - pad;
        int v = ox + kx - pad;
        int Hd = (H - 1) * dil + 1;
        int Wd = (W - 1) * dil + 1;
        if (u >= 0 && u < Hd && v >= 0 && v < Wd && (u % dil) == 0 && (v % dil) == 0) {
            val = (_Float16)X[(size_t)b * sN + (size_t)c * sC +
                              (size_t)(u / dil) * sH + (size_t)(v / dil) * sW];
        }
    }
    col[(size_t)mr * Kpad + k] = val;
}

// ---------------------------------------------------------------------------
// Deformable-conv im2col: col[mr][c*9+n] = mask_sigmoid * bilinear_sample.
// Matches reference deform_conv (1-based padded coords, clipping,
// g_lt/g_rb/g_lb/g_rt weights, sample n = ky*3+kx).
// ---------------------------------------------------------------------------
__global__ void k_deform_col(const float* __restrict__ X,    // NHWC, ch-stride Cs
                             const float* __restrict__ OFFb, // NHWC stride 32 (18 ch)
                             const float* __restrict__ MSKb, // NHWC stride 16 (9 ch)
                             _Float16* __restrict__ col,
                             int m0, int Mc, int H, int W, int Cs, int Cin, int Kpad)
{
    long long idx = (long long)blockIdx.x * blockDim.x + threadIdx.x;
    long long total = (long long)Mc * 9;
    if (idx >= total) return;
    int n  = (int)(idx % 9);
    int mr = (int)(idx / 9);
    int m  = m0 + mr;
    int j  = m % W;
    int t  = m / W;
    int i  = t % H;
    int b  = t / H;
    size_t pix = (size_t)m;
    float offx = OFFb[pix * 32 + n];
    float offy = OFFb[pix * 32 + 9 + n];
    float mval = 1.f / (1.f + expf(-MSKb[pix * 16 + n]));
    float pnx = (float)(n / 3 - 1);
    float pny = (float)(n % 3 - 1);
    int Hp = H + 2, Wp = W + 2;
    float px = fminf(fmaxf((float)(i + 1) + pnx + offx, 0.f), (float)(Hp - 1));
    float py = fminf(fmaxf((float)(j + 1) + pny + offy, 0.f), (float)(Wp - 1));
    float fx = floorf(px), fy = floorf(py);
    float x1f = fminf(fx + 1.f, (float)(Hp - 1));
    float y1f = fminf(fy + 1.f, (float)(Wp - 1));
    float glt = (1.f + (fx - px))  * (1.f + (fy - py));
    float grb = (1.f - (x1f - px)) * (1.f - (y1f - py));
    float glb = (1.f + (fx - px))  * (1.f - (y1f - py));
    float grt = (1.f - (x1f - px)) * (1.f + (fy - py));
    int u0 = (int)fx, v0 = (int)fy, u1 = (int)x1f, v1 = (int)y1f;
    const float* xb = X + (size_t)b * H * W * Cs;
    auto smp = [&](int u, int v, int c) -> float {
        if (u < 1 || u > H || v < 1 || v > W) return 0.f;
        return xb[((size_t)(u - 1) * W + (v - 1)) * Cs + c];
    };
    for (int c = 0; c < Cin; ++c) {
        float val = glt * smp(u0, v0, c) + grb * smp(u1, v1, c) +
                    glb * smp(u0, v1, c) + grt * smp(u1, v0, c);
        col[(size_t)mr * Kpad + c * 9 + n] = (_Float16)(val * mval);
    }
}

// ---------------------------------------------------------------------------
// WMMA GEMM: C[m][n] = sum_k A[m][k]*Bt[n][k] + bias[n].
// A: f16 [Mc][Kpad] row-major (streamed from global).
// Bt: f16 [Npad][Kpad] (B column-major). The 16 x Kpad B tile for this block
// is staged ONCE into LDS via GLOBAL_LOAD_ASYNC_TO_LDS_B128 (ASYNCcnt) and
// shared by all 4 waves; WMMA B fragments then come from ds_load_b128.
// One wave per 16x16 tile; K-loop in steps of 32 via v_wmma_f32_16x16x32_f16.
// Fragment layout per ISA: lane = (row = l&15, half = l>>4); elements are
// [row][k + half*8 + 0..7] and [row][k + 16 + half*8 + 0..7] (two b128s).
// REQUIRES: Mc % 64 == 0 (grid.x = Mc/64, 4 full waves per block).
// ---------------------------------------------------------------------------
__global__ void k_wmma_gemm(const _Float16* __restrict__ A,
                            const _Float16* __restrict__ Bt,
                            const float* __restrict__ bias, int biasN,
                            float* __restrict__ C,
                            int Mc, int Kpad, int Npad)
{
    __shared__ _Float16 btile[16 * KPAD_MAX];

    const int wid  = threadIdx.x >> 5;
    const int lane = threadIdx.x & 31;
    const int mt   = blockIdx.x * 4 + wid;
    const int nt   = blockIdx.y;
    const int row  = lane & 15;
    const int half = lane >> 4;

    // ---- async copy of the contiguous 16 x Kpad B tile into LDS ----
    {
        const _Float16* gsrc = Bt + (size_t)nt * 16 * Kpad;
        unsigned lbase = (unsigned)(unsigned long long)(&btile[0]);
        int nchunk = 2 * Kpad;                      // 16-byte chunks
        for (int c = threadIdx.x; c < nchunk; c += 128) {
            unsigned laddr = lbase + c * 16;
            const _Float16* gaddr = gsrc + c * 8;
            asm volatile("global_load_async_to_lds_b128 %0, %1, off"
                         :: "v"(laddr), "v"(gaddr) : "memory");
        }
        asm volatile("s_wait_asynccnt 0x0" ::: "memory");
        __syncthreads();
    }

    const _Float16* arow = A + (size_t)(mt * 16 + row) * Kpad + half * 8;
    const _Float16* brow = &btile[row * Kpad + half * 8];

    v8f acc = {0.f, 0.f, 0.f, 0.f, 0.f, 0.f, 0.f, 0.f};
    for (int k = 0; k < Kpad; k += 32) {
        __builtin_prefetch(arow + k + 32, 0, 1);    // speculative, next A chunk
        v8h a0 = *(const v8h*)(arow + k);
        v8h a1 = *(const v8h*)(arow + k + 16);
        v8h b0 = *(const v8h*)(brow + k);
        v8h b1 = *(const v8h*)(brow + k + 16);
        v16h a, b;
#pragma unroll
        for (int e = 0; e < 8; ++e) {
            a[e] = a0[e]; a[8 + e] = a1[e];
            b[e] = b0[e]; b[8 + e] = b1[e];
        }
        acc = __builtin_amdgcn_wmma_f32_16x16x32_f16(
            /*neg_a=*/false, a, /*neg_b=*/false, b,
            /*c_mod=*/(short)0, acc, /*reuse_a=*/false, /*reuse_b=*/false);
    }
    const int n0 = nt * 16 + row;
    const int m0 = mt * 16 + half * 8;
    const float bv = (n0 < biasN) ? bias[n0] : 0.f;
#pragma unroll
    for (int r = 0; r < 8; ++r)
        C[(size_t)(m0 + r) * Npad + n0] = acc[r] + bv;
}

// ---------------------------------------------------------------------------
// BatchNorm statistics: one block per channel, reduce over M pixels.
// ---------------------------------------------------------------------------
__global__ void k_bn_stats(const float* __restrict__ X, int M, int Cs,
                           float* __restrict__ mean, float* __restrict__ rstd,
                           float eps)
{
    __shared__ float s1[256], s2[256];
    int c = blockIdx.x;
    float a = 0.f, b = 0.f;
    for (int m = threadIdx.x; m < M; m += blockDim.x) {
        float v = X[(size_t)m * Cs + c];
        a += v; b += v * v;
    }
    s1[threadIdx.x] = a; s2[threadIdx.x] = b;
    __syncthreads();
    for (int st = 128; st > 0; st >>= 1) {
        if ((int)threadIdx.x < st) {
            s1[threadIdx.x] += s1[threadIdx.x + st];
            s2[threadIdx.x] += s2[threadIdx.x + st];
        }
        __syncthreads();
    }
    if (threadIdx.x == 0) {
        float mu  = s1[0] / (float)M;
        float var = s2[0] / (float)M - mu * mu;
        mean[c] = mu;
        rstd[c] = rsqrtf(var + eps);
    }
}

__global__ void k_bn_elu(float* __restrict__ X,
                         const float* __restrict__ mean, const float* __restrict__ rstd,
                         const float* __restrict__ g, const float* __restrict__ b,
                         long long total, int C)
{
    long long idx = (long long)blockIdx.x * blockDim.x + threadIdx.x;
    if (idx >= total) return;
    int c = (int)(idx % C);
    float y = g[c] * (X[idx] - mean[c]) * rstd[c] + b[c];
    X[idx] = y > 0.f ? y : expm1f(y);
}

__global__ void k_elu(float* __restrict__ X, long long total)
{
    long long idx = (long long)blockIdx.x * blockDim.x + threadIdx.x;
    if (idx >= total) return;
    float y = X[idx];
    X[idx] = y > 0.f ? y : expm1f(y);
}

// 2x2 average pool, NHWC (channel stride C), even H/W.
__global__ void k_avgpool2(const float* __restrict__ X, float* __restrict__ Y,
                           int Nb, int H, int W, int C)
{
    long long idx = (long long)blockIdx.x * blockDim.x + threadIdx.x;
    long long total = (long long)Nb * (H / 2) * (W / 2) * C;
    if (idx >= total) return;
    int c  = (int)(idx % C);
    long long t = idx / C;
    int ox = (int)(t % (W / 2)); t /= (W / 2);
    int oy = (int)(t % (H / 2));
    int b  = (int)(t / (H / 2));
    const float* p = X + (((size_t)b * H + 2 * oy) * W + 2 * ox) * C + c;
    Y[idx] = 0.25f * (p[0] + p[C] + p[(size_t)W * C] + p[(size_t)W * C + C]);
}

// ---------------------------------------------------------------------------
// Final: V = dec transposed (V[b,i,j,c] = dec[b,j,i,c]); warp_bicubic(V, x1[:,0]).
// dec is NHWC f32, 28x28, channel stride 16 (channels 0,1 real).
// ---------------------------------------------------------------------------
__device__ __forceinline__ float cubicf(float A, float Bv, float Cc, float D, float t)
{
    float a = -0.5f * A + 1.5f * Bv - 1.5f * Cc + 0.5f * D;
    float b =  A - 2.5f * Bv + 2.0f * Cc - 0.5f * D;
    float c = -0.5f * A + 0.5f * Cc;
    return ((a * t + b) * t + c) * t + Bv;
}

__global__ void k_warp(const float* __restrict__ dec, const float* __restrict__ x1,
                       float* __restrict__ out)
{
    long long idx = (long long)blockIdx.x * blockDim.x + threadIdx.x;
    long long total = 256LL * 28 * 28;
    if (idx >= total) return;
    int j = (int)(idx % 28);
    long long t = idx / 28;
    int i = (int)(t % 28);
    int b = (int)(t / 28);
    // V[b,i,j,:] = dec[b, j, i, :]
    const float* dv = dec + (((size_t)b * 28 + j) * 28 + i) * 16;
    float gx = (-1.f + 2.f * (float)j / 27.f) + dv[0];
    float gy = (-1.f + 2.f * (float)i / 27.f) + dv[1];
    float ix = (gx + 1.f) * 13.5f;   // *(w-1)/2
    float iy = (gy + 1.f) * 13.5f;
    float x0f = floorf(ix), y0f = floorf(iy);
    float tx = ix - x0f, ty = iy - y0f;
    const float* img = x1 + (size_t)b * 2 * 784;   // x1[b,0]
    auto g = [&](float yy, float xx) -> float {
        int yi = (int)fminf(fmaxf(yy, 0.f), 27.f);
        int xi = (int)fminf(fmaxf(xx, 0.f), 27.f);
        return img[yi * 28 + xi];
    };
    float rows[4];
#pragma unroll
    for (int ky = -1; ky <= 2; ++ky) {
        float c0 = g(y0f + ky, x0f - 1.f);
        float c1 = g(y0f + ky, x0f);
        float c2 = g(y0f + ky, x0f + 1.f);
        float c3 = g(y0f + ky, x0f + 2.f);
        rows[ky + 1] = cubicf(c0, c1, c2, c3, tx);
    }
    out[idx] = cubicf(rows[0], rows[1], rows[2], rows[3], ty);
}

// ===========================================================================
// Host-side drivers
// ===========================================================================
static void prep_w(hipStream_t s, const float* w, _Float16* Bt,
                   int Cout, int Cin, int KH, int KW, int Kpad, int Npad, int tr)
{
    long long tot = (long long)Npad * Kpad;
    k_prep_wB<<<dim3((unsigned)((tot + 255) / 256)), 256, 0, s>>>(
        w, Bt, Cout, Cin, KH, KW, Kpad, Npad, tr);
}

static void run_conv(hipStream_t s,
                     const float* X, long long sN, long long sC, long long sH, long long sW,
                     int Nb, int H, int W, int Cin, int KH, int KW, int pad, int dil,
                     const _Float16* Bt, int Npad, const float* bias, int biasN,
                     _Float16* col, float* Y,
                     const _Float16* Bt2 = nullptr, int Npad2 = 0,
                     const float* bias2 = nullptr, int biasN2 = 0, float* Y2 = nullptr)
{
    int Hd = (H - 1) * dil + 1, Wd = (W - 1) * dil + 1;
    int OH = Hd + 2 * pad - KH + 1, OW = Wd + 2 * pad - KW + 1;
    int M = Nb * OH * OW;
    int K = Cin * KH * KW;
    int Kpad = (K + 31) & ~31;
    for (int m0 = 0; m0 < M; m0 += CHUNK_M) {
        int Mc = (M - m0 < CHUNK_M) ? (M - m0) : CHUNK_M;   // always % 64 == 0 here
        long long tot = (long long)Mc * Kpad;
        k_im2col<<<dim3((unsigned)((tot + 255) / 256)), 256, 0, s>>>(
            X, col, m0, Mc, H, W, Cin, sN, sC, sH, sW, OH, OW, KH, KW, pad, dil, Kpad);
        dim3 grd((unsigned)(Mc / 64), (unsigned)(Npad / 16));
        k_wmma_gemm<<<grd, 128, 0, s>>>(col, Bt, bias, biasN,
                                        Y + (size_t)m0 * Npad, Mc, Kpad, Npad);
        if (Bt2) {
            dim3 grd2(grd.x, (unsigned)(Npad2 / 16));
            k_wmma_gemm<<<grd2, 128, 0, s>>>(col, Bt2, bias2, biasN2,
                                             Y2 + (size_t)m0 * Npad2, Mc, Kpad, Npad2);
        }
    }
}

static void run_deform(hipStream_t s, const float* X, int Nb, int H, int W, int Cs, int Cin,
                       const float* OFFb, const float* MSKb,
                       const _Float16* Bt, int Npad, _Float16* col, float* Y)
{
    int M = Nb * H * W;
    int Kpad = Cin * 9;  // 576 / 1152: multiples of 32
    for (int m0 = 0; m0 < M; m0 += CHUNK_M) {
        int Mc = (M - m0 < CHUNK_M) ? (M - m0) : CHUNK_M;   // always % 64 == 0 here
        long long tot = (long long)Mc * 9;
        k_deform_col<<<dim3((unsigned)((tot + 255) / 256)), 256, 0, s>>>(
            X, OFFb, MSKb, col, m0, Mc, H, W, Cs, Cin, Kpad);
        dim3 grd((unsigned)(Mc / 64), (unsigned)(Npad / 16));
        k_wmma_gemm<<<grd, 128, 0, s>>>(col, Bt, nullptr, 0,
                                        Y + (size_t)m0 * Npad, Mc, Kpad, Npad);
    }
}

static void run_bn_elu(hipStream_t s, float* X, int M, int C,
                       const float* g, const float* b, float* mean, float* rstd)
{
    k_bn_stats<<<dim3((unsigned)C), 256, 0, s>>>(X, M, C, mean, rstd, 1e-5f);
    long long tot = (long long)M * C;
    k_bn_elu<<<dim3((unsigned)((tot + 255) / 256)), 256, 0, s>>>(X, mean, rstd, g, b, tot, C);
}

extern "C" void kernel_launch(void* const* d_in, const int* in_sizes, int n_in,
                              void* d_out, int out_size, void* d_ws, size_t ws_size,
                              hipStream_t stream)
{
    (void)in_sizes; (void)n_in; (void)out_size; (void)ws_size;
    const float* x1       = (const float*)d_in[0];
    const float* conv1_w  = (const float*)d_in[9];
    const float* conv1_b  = (const float*)d_in[10];
    const float* bn1_g    = (const float*)d_in[11];
    const float* bn1_b    = (const float*)d_in[12];
    const float* dc2_pw   = (const float*)d_in[13];
    const float* dc2_pb   = (const float*)d_in[14];
    const float* dc2_mw   = (const float*)d_in[15];
    const float* dc2_mb   = (const float*)d_in[16];
    const float* dc2_w    = (const float*)d_in[17];
    const float* bn2_g    = (const float*)d_in[18];
    const float* bn2_b    = (const float*)d_in[19];
    const float* conv3_w  = (const float*)d_in[20];
    const float* conv3_b  = (const float*)d_in[21];
    const float* bn3_g    = (const float*)d_in[22];
    const float* bn3_b    = (const float*)d_in[23];
    const float* dc41_pw  = (const float*)d_in[24];
    const float* dc41_pb  = (const float*)d_in[25];
    const float* dc41_mw  = (const float*)d_in[26];
    const float* dc41_mb  = (const float*)d_in[27];
    const float* dc41_w   = (const float*)d_in[28];
    const float* bn41_g   = (const float*)d_in[29];
    const float* bn41_b   = (const float*)d_in[30];
    const float* conv4_w  = (const float*)d_in[31];
    const float* conv4_b  = (const float*)d_in[32];
    const float* conv4m_w = (const float*)d_in[33];
    const float* conv4m_b = (const float*)d_in[34];
    const float* conv3m_w = (const float*)d_in[35];
    const float* conv3m_b = (const float*)d_in[36];
    const float* conv2m_w = (const float*)d_in[37];
    const float* conv2m_b = (const float*)d_in[38];
    const float* conv1m_w = (const float*)d_in[39];
    const float* conv1m_b = (const float*)d_in[40];

    char* ws = (char*)d_ws;
    const size_t MB = 1024 * 1024;
    auto F16 = [&](size_t off) { return (_Float16*)(ws + off); };
    auto F32 = [&](size_t off) { return (float*)(ws + off); };

    // f16 weight buffers (region [0, 1MB))
    _Float16* wb_conv1 = F16(0);        // 64 x 32
    _Float16* wb_dc2p  = F16(4096);     // 32 x 576
    _Float16* wb_dc2m  = F16(40960);    // 16 x 576
    _Float16* wb_dc2w  = F16(59392);    // 128 x 576
    _Float16* wb_conv3 = F16(206848);   // 128 x 1152
    _Float16* wb_dc41p = F16(501760);   // 32 x 1152
    _Float16* wb_dc41m = F16(575488);   // 16 x 1152
    _Float16* wb_dc41w = F16(612352);   // 64 x 1152
    _Float16* wb_conv4 = F16(759808);   // 16 x 576
    _Float16* wb_c4m   = F16(778240);   // 128 x 32
    _Float16* wb_c3m   = F16(786432);   // 64 x 1152
    _Float16* wb_c2m   = F16(933888);   // 32 x 576
    _Float16* wb_c1m   = F16(970752);   // 16 x 128

    float* st = F32(1 * MB);            // BN stats scratch
    float *mean1 = st,        *rstd1 = st + 256;
    float *mean2 = st + 512,  *rstd2 = st + 768;
    float *mean3 = st + 1024, *rstd3 = st + 1280;
    float *mean4 = st + 1536, *rstd4 = st + 1792;

    _Float16* COL = F16(2 * MB);        // im2col chunk: 16384 x 1152 f16 = 36MB

    float* A0   = F32(40 * MB);   // conv1 out   200704 x 64
    float* A1   = F32(92 * MB);   // pool1       50176 x 64
    float* OFF2 = F32(105 * MB);  // offsets     50176 x 32
    float* MSK2 = F32(112 * MB);  // mask        50176 x 16
    float* A3   = F32(116 * MB);  // dc2 out     50176 x 128
    float* A4   = F32(142 * MB);  // conv3 out   50176 x 128
    float* A5   = F32(168 * MB);  // pool2       12544 x 128
    float* OFF4 = F32(175 * MB);  // offsets     12544 x 32
    float* MSK4 = F32(177 * MB);  // mask        12544 x 16
    float* A7   = F32(178 * MB);  // dc41 out    12544 x 64
    float* A8   = F32(182 * MB);  // conv4 out   12544 x 16
    float* A9   = F32(183 * MB);  // conv4m out  43264 x 128
    float* A10  = F32(206 * MB);  // conv3m out  57600 x 64
    float* A11  = F32(221 * MB);  // conv2m out  215296 x 32
    float* A12  = F32(249 * MB);  // conv1m out  200704 x 16

    // --- weight conversion -------------------------------------------------
    prep_w(stream, conv1_w,  wb_conv1, 64,  2,   3, 3, 32,   64,  0);
    prep_w(stream, dc2_pw,   wb_dc2p,  18,  64,  3, 3, 576,  32,  0);
    prep_w(stream, dc2_mw,   wb_dc2m,  9,   64,  3, 3, 576,  16,  0);
    prep_w(stream, dc2_w,    wb_dc2w,  128, 64,  3, 3, 576,  128, 0);
    prep_w(stream, conv3_w,  wb_conv3, 128, 128, 3, 3, 1152, 128, 0);
    prep_w(stream, dc41_pw,  wb_dc41p, 18,  128, 3, 3, 1152, 32,  0);
    prep_w(stream, dc41_mw,  wb_dc41m, 9,   128, 3, 3, 1152, 16,  0);
    prep_w(stream, dc41_w,   wb_dc41w, 64,  128, 3, 3, 1152, 64,  0);
    prep_w(stream, conv4_w,  wb_conv4, 2,   64,  3, 3, 576,  16,  0);
    prep_w(stream, conv4m_w, wb_c4m,   128, 2,   3, 3, 32,   128, 1);
    prep_w(stream, conv3m_w, wb_c3m,   64,  128, 3, 3, 1152, 64,  1);
    prep_w(stream, conv2m_w, wb_c2m,   32,  64,  3, 3, 576,  32,  1);
    prep_w(stream, conv1m_w, wb_c1m,   2,   32,  2, 2, 128,  16,  1);

    // --- encoder -----------------------------------------------------------
    // conv1 (x1 is NCHW: sN=2*784, sC=784, sH=28, sW=1) -> A0 (NHWC c64)
    run_conv(stream, x1, 1568, 784, 28, 1, 256, 28, 28, 2, 3, 3, 1, 1,
             wb_conv1, 64, conv1_b, 64, COL, A0);
    run_bn_elu(stream, A0, 200704, 64, bn1_g, bn1_b, mean1, rstd1);
    {
        long long tot = 50176LL * 64;
        k_avgpool2<<<dim3((unsigned)((tot + 255) / 256)), 256, 0, stream>>>(A0, A1, 256, 28, 28, 64);
    }
    // deform block 2: offset & mask convs share one im2col
    run_conv(stream, A1, 14 * 14 * 64, 1, 14 * 64, 64, 256, 14, 14, 64, 3, 3, 1, 1,
             wb_dc2p, 32, dc2_pb, 18, COL, OFF2, wb_dc2m, 16, dc2_mb, 9, MSK2);
    run_deform(stream, A1, 256, 14, 14, 64, 64, OFF2, MSK2, wb_dc2w, 128, COL, A3);
    run_bn_elu(stream, A3, 50176, 128, bn2_g, bn2_b, mean2, rstd2);
    // conv3
    run_conv(stream, A3, 14 * 14 * 128, 1, 14 * 128, 128, 256, 14, 14, 128, 3, 3, 1, 1,
             wb_conv3, 128, conv3_b, 128, COL, A4);
    run_bn_elu(stream, A4, 50176, 128, bn3_g, bn3_b, mean3, rstd3);
    {
        long long tot = 12544LL * 128;
        k_avgpool2<<<dim3((unsigned)((tot + 255) / 256)), 256, 0, stream>>>(A4, A5, 256, 14, 14, 128);
    }
    // deform block 41
    run_conv(stream, A5, 49 * 128, 1, 7 * 128, 128, 256, 7, 7, 128, 3, 3, 1, 1,
             wb_dc41p, 32, dc41_pb, 18, COL, OFF4, wb_dc41m, 16, dc41_mb, 9, MSK4);
    run_deform(stream, A5, 256, 7, 7, 128, 128, OFF4, MSK4, wb_dc41w, 64, COL, A7);
    run_bn_elu(stream, A7, 12544, 64, bn41_g, bn41_b, mean4, rstd4);
    // conv4 (7x7 bicubic resize afterwards is identity)
    run_conv(stream, A7, 49 * 64, 1, 7 * 64, 64, 256, 7, 7, 64, 3, 3, 1, 1,
             wb_conv4, 16, conv4_b, 2, COL, A8);

    // --- decoder (transposed convs via lhs-dilated im2col) -------------------
    // conv4m: stride2 pad1 k3  ->  dil=2, p=1, 7->13
    run_conv(stream, A8, 49 * 16, 1, 7 * 16, 16, 256, 7, 7, 2, 3, 3, 1, 2,
             wb_c4m, 128, conv4m_b, 128, COL, A9);
    // conv3m: stride1 pad0 k3  ->  dil=1, p=2, 13->15, then elu
    run_conv(stream, A9, 169 * 128, 1, 13 * 128, 128, 256, 13, 13, 128, 3, 3, 2, 1,
             wb_c3m, 64, conv3m_b, 64, COL, A10);
    {
        long long tot = 57600LL * 64;
        k_elu<<<dim3((unsigned)((tot + 255) / 256)), 256, 0, stream>>>(A10, tot);
    }
    // conv2m: stride2 pad1 k3  ->  dil=2, p=1, 15->29, then elu
    run_conv(stream, A10, 225 * 64, 1, 15 * 64, 64, 256, 15, 15, 64, 3, 3, 1, 2,
             wb_c2m, 32, conv2m_b, 32, COL, A11);
    {
        long long tot = 215296LL * 32;
        k_elu<<<dim3((unsigned)((tot + 255) / 256)), 256, 0, stream>>>(A11, tot);
    }
    // conv1m: stride1 pad1 k2  ->  dil=1, p=0, 29->28, then elu
    run_conv(stream, A11, 841 * 32, 1, 29 * 32, 32, 256, 29, 29, 32, 2, 2, 0, 1,
             wb_c1m, 16, conv1m_b, 2, COL, A12);
    {
        long long tot = 200704LL * 16;
        k_elu<<<dim3((unsigned)((tot + 255) / 256)), 256, 0, stream>>>(A12, tot);
    }

    // --- final warp (28x28 bicubic resize is identity; only transpose) ------
    {
        long long tot = 256LL * 28 * 28;
        k_warp<<<dim3((unsigned)((tot + 255) / 256)), 256, 0, stream>>>(A12, x1, (float*)d_out);
    }
}
